// LSTM_scratch_38603166056484
// MI455X (gfx1250) — compile-verified, matrix operand order checked by reading, not run on hardware
//
#include <hip/hip_runtime.h>
#include <hip/hip_bf16.h>

// ---------------------------------------------------------------------------
// CDNA5 (gfx1250, wave32) WMMA types
// ---------------------------------------------------------------------------
typedef __attribute__((ext_vector_type(16))) __bf16 v16bf;
typedef __attribute__((ext_vector_type(8)))  __bf16 v8bf;
typedef __attribute__((ext_vector_type(8)))  float  v8f;

#define B_SZ    128
#define HID     1024
#define GATES4  4096
#define T_STEPS 256
#define IN_DIM  256
#define OUT_DIM 10
#define NBLOCKS 64            // persistent grid: one block per 16 hidden units

// ---------------------------------------------------------------------------
// Fragment helpers
// ---------------------------------------------------------------------------
// 16-bit A (16x32 MxK) layout (ISA 7.12.2): lane = m + 16*kh; per lane the 16
// bf16 are two contiguous K-runs of 8:  K = 8*kh + j  and  K = 16 + 8*kh + j.
__device__ __forceinline__ v16bf load_a_frag(const __bf16* __restrict__ base,
                                             int rowStride, int m0, int k0,
                                             int lane) {
    int m  = m0 + (lane & 15);
    int kh = lane >> 4;
    const __bf16* p = base + (size_t)m * rowStride + k0 + 8 * kh;
    v8bf lo = *(const v8bf*)p;          // 16B aligned
    v8bf hi = *(const v8bf*)(p + 16);   // 16B aligned
    v16bf a;
#pragma unroll
    for (int j = 0; j < 8; ++j) { a[j] = lo[j]; a[j + 8] = hi[j]; }
    return a;
}

__device__ __forceinline__ v8f wmma_bf16(v16bf a, v16bf b, v8f c) {
    return __builtin_amdgcn_wmma_f32_16x16x32_bf16(
        false, a, false, b, (short)0, c, false, false);
}

__device__ __forceinline__ float sigf(float x) {
    return 1.0f / (1.0f + __expf(-x));
}

// K-index map used by the weight packer (same ordering as load_a_frag).
__device__ __forceinline__ int kmap(int j, int kh) {
    return (j < 8) ? (8 * kh + j) : (16 + 8 * kh + (j - 8));
}

// ---------------------------------------------------------------------------
// Pre-processing kernels (one-time per call)
// ---------------------------------------------------------------------------
// Pack fp32 W[Ktot x 4096] -> bf16 B fragments. Fragment f = ntile*nKtiles+kt,
// layout P[f*512 + lane*16 + j]; lane holds column n = ntile*16 + lane%16.
__global__ void pack_weight(const float* __restrict__ W, __bf16* __restrict__ P,
                            int nKtiles) {
    int gid = blockIdx.x * blockDim.x + threadIdx.x;
    if (gid >= 256 * nKtiles * 32) return;
    int fid  = gid >> 5;
    int lane = gid & 31;
    int ntile = fid / nKtiles;
    int ktile = fid % nKtiles;
    int n  = ntile * 16 + (lane & 15);
    int kh = lane >> 4;
    __bf16* out = P + (size_t)fid * 512 + lane * 16;
#pragma unroll
    for (int j = 0; j < 16; ++j)
        out[j] = (__bf16)W[(size_t)(ktile * 32 + kmap(j, kh)) * GATES4 + n];
}

__global__ void bias_fold(const float* __restrict__ bi,
                          const float* __restrict__ bh,
                          float* __restrict__ out) {
    int i = blockIdx.x * blockDim.x + threadIdx.x;
    if (i < GATES4) out[i] = bi[i] + bh[i];
}

__global__ void cvt_f32_bf16(const float* __restrict__ src,
                             __bf16* __restrict__ dst, int n) {
    int i = blockIdx.x * blockDim.x + threadIdx.x;
    if (i < n) dst[i] = (__bf16)src[i];
}

// ---------------------------------------------------------------------------
// Grid-wide barrier: monotonically increasing counter, per-barrier goal.
// ---------------------------------------------------------------------------
__device__ __forceinline__ void grid_barrier(unsigned* cnt, unsigned goal) {
    __syncthreads();
    if (threadIdx.x == 0) {
        __threadfence();                 // release our h stores to device scope
        atomicAdd(cnt, 1u);
        while (__hip_atomic_load(cnt, __ATOMIC_RELAXED,
                                 __HIP_MEMORY_SCOPE_AGENT) < goal) {
            __builtin_amdgcn_s_sleep(1);
        }
        __threadfence();                 // acquire: invalidate stale lines
    }
    __syncthreads();
}

// ---------------------------------------------------------------------------
// One LSTM layer step for one wave: z tiles for ALL 4 gates of
// (16 batch rows x 16 hidden cols), then in-register gate fusion.
// cv[8] = cell state (f32, lives in registers across all 256 steps).
// ---------------------------------------------------------------------------
__device__ __forceinline__ void layer_step(
    const __bf16* __restrict__ X, int xStride, int xKtiles,
    const __bf16* __restrict__ Hin,
    const __bf16* const* px,      // 4 packed-Wi fragment streams (per gate)
    const __bf16* const* ph,      // 4 packed-Wh fragment streams (per gate)
    const float* b,               // 4 folded biases for this lane's column
    float* cv,                    // 8 cell values
    __bf16* __restrict__ Hout,
    int mtile, int lane, int col) {
    v8f acc[4] = {{}, {}, {}, {}};
    for (int kt = 0; kt < xKtiles; ++kt) {
        v16bf a = load_a_frag(X, xStride, mtile * 16, kt * 32, lane);
#pragma unroll
        for (int g = 0; g < 4; ++g) {
            v16bf bb = *(const v16bf*)(px[g] + (size_t)kt * 512);
            acc[g] = wmma_bf16(a, bb, acc[g]);
        }
    }
    for (int kt = 0; kt < 32; ++kt) {
        v16bf a = load_a_frag(Hin, HID, mtile * 16, kt * 32, lane);
#pragma unroll
        for (int g = 0; g < 4; ++g) {
            v16bf bb = *(const v16bf*)(ph[g] + (size_t)kt * 512);
            acc[g] = wmma_bf16(a, bb, acc[g]);
        }
    }
    // D layout: lane -> n = lane%16, m = r + 8*(lane/16)
    const int mhalf = lane >> 4;
#pragma unroll
    for (int r = 0; r < 8; ++r) {
        float f  = sigf(acc[0][r] + b[0]);
        float ii = sigf(acc[1][r] + b[1]);
        float gg = tanhf(acc[2][r] + b[2]);
        float oo = sigf(acc[3][r] + b[3]);
        cv[r] = cv[r] * f + ii * gg;
        int m = mtile * 16 + mhalf * 8 + r;
        Hout[(size_t)m * HID + col] = (__bf16)(tanhf(cv[r]) * oo);
    }
}

// ---------------------------------------------------------------------------
// Persistent kernel: entire 2-layer, 256-step recurrence + classifier.
// Grid: 64 blocks x 256 threads (8 waves). Wave w = mtile (16 batch rows),
// block owns 16 hidden columns for both layers.
// ---------------------------------------------------------------------------
__global__ __launch_bounds__(256, 1)
void lstm_persistent(const __bf16* __restrict__ xbf,
                     const __bf16* __restrict__ pWi1,
                     const __bf16* __restrict__ pWh1,
                     const __bf16* __restrict__ pWi2,
                     const __bf16* __restrict__ pWh2,
                     const float* __restrict__ bias1,
                     const float* __restrict__ bias2,
                     const float* __restrict__ c01,
                     const float* __restrict__ c02,
                     __bf16* __restrict__ h1a, __bf16* __restrict__ h1b,
                     __bf16* __restrict__ h2a, __bf16* __restrict__ h2b,
                     const float* __restrict__ Wc,
                     const float* __restrict__ bc,
                     float* __restrict__ y,
                     unsigned* __restrict__ barCnt) {
    const int tid   = threadIdx.x;
    const int lane  = tid & 31;
    const int mtile = tid >> 5;              // 0..7
    const int nblk  = blockIdx.x;            // 0..63
    const int n0    = lane & 15;
    const int mhalf = lane >> 4;
    const int col   = nblk * 16 + n0;

    // Per-gate packed-weight fragment streams for this wave.
    const __bf16* px1[4]; const __bf16* ph1[4];
    const __bf16* px2[4]; const __bf16* ph2[4];
    float b1[4], b2[4];
#pragma unroll
    for (int g = 0; g < 4; ++g) {
        size_t ct = (size_t)(g * 64 + nblk);
        px1[g] = pWi1 + ct * 8  * 512 + lane * 16;
        ph1[g] = pWh1 + ct * 32 * 512 + lane * 16;
        px2[g] = pWi2 + ct * 32 * 512 + lane * 16;
        ph2[g] = pWh2 + ct * 32 * 512 + lane * 16;
        b1[g] = bias1[g * HID + col];
        b2[g] = bias2[g * HID + col];
    }

    // Cell state in registers for the whole recurrence.
    float cv1[8], cv2[8];
#pragma unroll
    for (int r = 0; r < 8; ++r) {
        int m = mtile * 16 + mhalf * 8 + r;
        cv1[r] = c01[(size_t)m * HID + col];
        cv2[r] = c02[(size_t)m * HID + col];
    }

    __bf16* h1buf[2] = {h1a, h1b};
    __bf16* h2buf[2] = {h2a, h2b};

    unsigned goal = 0;
    for (int t = 0; t < T_STEPS; ++t) {
        int cur = t & 1, nxt = 1 - cur;
        // layer 1: x_t is row-contiguous bf16, row stride T*IN_DIM
        layer_step(xbf + (size_t)t * IN_DIM, T_STEPS * IN_DIM, 8,
                   h1buf[cur], px1, ph1, b1, cv1, h1buf[nxt],
                   mtile, lane, col);
        goal += NBLOCKS; grid_barrier(barCnt, goal);
        // layer 2: input is layer-1's fresh hidden state
        layer_step(h1buf[nxt], HID, 32,
                   h2buf[cur], px2, ph2, b2, cv2, h2buf[nxt],
                   mtile, lane, col);
        goal += NBLOCKS; grid_barrier(barCnt, goal);
    }

    // Classifier: y = h2_final @ Wc + bc  (h2 final buffer = index 0, T even)
    const __bf16* hfin = h2buf[1 - ((T_STEPS - 1) & 1)];
    int gid = nblk * 256 + tid;
    if (gid < B_SZ * OUT_DIM) {
        int b = gid / OUT_DIM, o = gid % OUT_DIM;
        const __bf16* h = hfin + (size_t)b * HID;
        float s = bc[o];
        for (int k = 0; k < HID; ++k) s += (float)h[k] * Wc[k * OUT_DIM + o];
        y[gid] = s;
    }
}

// ---------------------------------------------------------------------------
// Host driver
// ---------------------------------------------------------------------------
extern "C" void kernel_launch(void* const* d_in, const int* in_sizes, int n_in,
                              void* d_out, int out_size, void* d_ws, size_t ws_size,
                              hipStream_t stream) {
    const float* x   = (const float*)d_in[0];
    const float* h01 = (const float*)d_in[1];
    const float* c01 = (const float*)d_in[2];
    const float* h02 = (const float*)d_in[3];
    const float* c02 = (const float*)d_in[4];
    const float* Wi1 = (const float*)d_in[5];
    const float* Wh1 = (const float*)d_in[6];
    const float* bi1 = (const float*)d_in[7];
    const float* bh1 = (const float*)d_in[8];
    const float* Wi2 = (const float*)d_in[9];
    const float* Wh2 = (const float*)d_in[10];
    const float* bi2 = (const float*)d_in[11];
    const float* bh2 = (const float*)d_in[12];
    const float* Wc  = (const float*)d_in[13];
    const float* bc  = (const float*)d_in[14];
    float* y = (float*)d_out;

    // ---- workspace carve-up -------------------------------------------
    char* ws = (char*)d_ws;
    size_t off = 0;
    auto carve = [&](size_t bytes) {
        void* p = ws + off;
        off += (bytes + 1023) & ~(size_t)1023;
        return p;
    };
    const size_t FRAG = 512 * sizeof(__bf16);            // 1 KB / fragment
    __bf16* pWi1 = (__bf16*)carve(256 * 8  * FRAG);
    __bf16* pWh1 = (__bf16*)carve(256 * 32 * FRAG);
    __bf16* pWi2 = (__bf16*)carve(256 * 32 * FRAG);
    __bf16* pWh2 = (__bf16*)carve(256 * 32 * FRAG);
    float*  bias1 = (float*)carve(GATES4 * sizeof(float));
    float*  bias2 = (float*)carve(GATES4 * sizeof(float));
    __bf16* xbf   = (__bf16*)carve((size_t)B_SZ * T_STEPS * IN_DIM * sizeof(__bf16));
    __bf16* h1a   = (__bf16*)carve((size_t)B_SZ * HID * sizeof(__bf16));
    __bf16* h1b   = (__bf16*)carve((size_t)B_SZ * HID * sizeof(__bf16));
    __bf16* h2a   = (__bf16*)carve((size_t)B_SZ * HID * sizeof(__bf16));
    __bf16* h2b   = (__bf16*)carve((size_t)B_SZ * HID * sizeof(__bf16));
    unsigned* barCnt = (unsigned*)carve(1024);

    // ---- one-time preprocessing (deterministic, capture-safe) ---------
    hipMemsetAsync(barCnt, 0, 1024, stream);

    auto packBlocks = [](int nKtiles) { return (256 * nKtiles * 32 + 255) / 256; };
    pack_weight<<<packBlocks(8),  256, 0, stream>>>(Wi1, pWi1, 8);
    pack_weight<<<packBlocks(32), 256, 0, stream>>>(Wh1, pWh1, 32);
    pack_weight<<<packBlocks(32), 256, 0, stream>>>(Wi2, pWi2, 32);
    pack_weight<<<packBlocks(32), 256, 0, stream>>>(Wh2, pWh2, 32);
    bias_fold<<<GATES4 / 256, 256, 0, stream>>>(bi1, bh1, bias1);
    bias_fold<<<GATES4 / 256, 256, 0, stream>>>(bi2, bh2, bias2);

    const int nx = B_SZ * T_STEPS * IN_DIM;
    cvt_f32_bf16<<<(nx + 255) / 256, 256, 0, stream>>>(x, xbf, nx);
    const int nh = B_SZ * HID;
    cvt_f32_bf16<<<(nh + 255) / 256, 256, 0, stream>>>(h01, h1a, nh);
    cvt_f32_bf16<<<(nh + 255) / 256, 256, 0, stream>>>(h02, h2a, nh);

    // ---- entire recurrence + classifier in one persistent kernel ------
    lstm_persistent<<<NBLOCKS, 256, 0, stream>>>(
        xbf, pWi1, pWh1, pWi2, pWh2, bias1, bias2, c01, c02,
        h1a, h1b, h2a, h2b, Wc, bc, y, barCnt);
}
